// SpikeFP16Linear_MultiPrecision_11450382811512
// MI455X (gfx1250) — compile-verified, matrix operand order checked by reading, not run on hardware
//
#include <hip/hip_runtime.h>

typedef __attribute__((ext_vector_type(16))) _Float16 v16h;
typedef __attribute__((ext_vector_type(8)))  _Float16 v8h;
typedef __attribute__((ext_vector_type(8)))  float    v8f;
typedef __attribute__((ext_vector_type(4)))  float    v4f;

#define K_DIM   2048
#define N_DIM   2048
#define BM      128
#define BN      128
#define BK      32
#define LDSP    40                 // padded LDS row stride in halves (32 + 8 pad)
#define K_ITERS (K_DIM / BK)

// Y = X @ W^T ; X:[M,K] fp32, W:[N,K] fp32. Output: fp16(Y) expanded to 16
// float32 bit-pulses per element, MSB first: out[m][n][j] = bit(15-j).
__global__ __launch_bounds__(256)
void spike_fp16_gemm_pulse(const float* __restrict__ X,
                           const float* __restrict__ W,
                           float* __restrict__ out) {
  __shared__ _Float16 ldsA[2][BM * LDSP];
  __shared__ _Float16 ldsB[2][BN * LDSP];

  const int t     = threadIdx.x;
  const int lane  = t & 31;
  const int wave  = t >> 5;
  const int lhalf = lane & 15;     // row within 16x16 operand tile
  const int khi   = lane >> 4;     // K-half select (0/1) per WMMA A/B layout
  const int wm    = wave & 3;      // wave M sub-tile: *32
  const int wn    = wave >> 2;     // wave N sub-tile: *64

  const int bm = blockIdx.x * BM;
  const int bn = blockIdx.y * BN;

  // Cooperative tile loader: thread t covers row t/2, cols (t&1)*16 .. +15
  const int lrow = t >> 1;
  const int lcol = (t & 1) << 4;
  const int lofs = lrow * LDSP + lcol;
  const float* xg = X + (size_t)(bm + lrow) * K_DIM + lcol;
  const float* wg = W + (size_t)(bn + lrow) * K_DIM + lcol;

  v4f ax[4], bx[4];                // prefetch registers (16 fp32 each operand)

  auto gload = [&](int kk) {
    const v4f* xp = (const v4f*)(xg + kk);
    const v4f* wp = (const v4f*)(wg + kk);
#pragma unroll
    for (int i = 0; i < 4; ++i) { ax[i] = xp[i]; bx[i] = wp[i]; }
  };

  // fp32 -> fp16 (RNE, matches astype(float16)) and stash into LDS stage
  auto cvt_store = [&](int stage) {
    v8h h0, h1, g0, g1;
#pragma unroll
    for (int i = 0; i < 2; ++i) {
#pragma unroll
      for (int j = 0; j < 4; ++j) {
        h0[4 * i + j] = (_Float16)ax[i][j];
        h1[4 * i + j] = (_Float16)ax[i + 2][j];
        g0[4 * i + j] = (_Float16)bx[i][j];
        g1[4 * i + j] = (_Float16)bx[i + 2][j];
      }
    }
    *(v8h*)&ldsA[stage][lofs]     = h0;
    *(v8h*)&ldsA[stage][lofs + 8] = h1;
    *(v8h*)&ldsB[stage][lofs]     = g0;
    *(v8h*)&ldsB[stage][lofs + 8] = g1;
  };

  const int kb = khi * 8;          // K sub-offset per WMMA 16-bit operand layout

  v8f acc[2][4] = {};

  gload(0);
  cvt_store(0);

#pragma unroll 2
  for (int it = 0; it < K_ITERS; ++it) {
    __syncthreads();
    const int cur = it & 1;

    if (it + 1 < K_ITERS) gload((it + 1) * BK);

    // Fetch WMMA operands from LDS.
    // Lane layout (16-bit A/B, wave32): lanes 0-15 -> K = kb..kb+7 and
    // kb+16..kb+23; lanes 16-31 -> kb=8 variant. Two ds_load_b128 each.
    v16h a[2], b[4];
#pragma unroll
    for (int im = 0; im < 2; ++im) {
      const _Float16* p = &ldsA[cur][(wm * 32 + im * 16 + lhalf) * LDSP + kb];
      v8h lo = *(const v8h*)p;
      v8h hi = *(const v8h*)(p + 16);
      a[im] = __builtin_shufflevector(lo, hi, 0, 1, 2, 3, 4, 5, 6, 7,
                                      8, 9, 10, 11, 12, 13, 14, 15);
    }
#pragma unroll
    for (int in = 0; in < 4; ++in) {
      const _Float16* p = &ldsB[cur][(wn * 64 + in * 16 + lhalf) * LDSP + kb];
      v8h lo = *(const v8h*)p;
      v8h hi = *(const v8h*)(p + 16);
      b[in] = __builtin_shufflevector(lo, hi, 0, 1, 2, 3, 4, 5, 6, 7,
                                      8, 9, 10, 11, 12, 13, 14, 15);
    }

#pragma unroll
    for (int im = 0; im < 2; ++im)
#pragma unroll
      for (int in = 0; in < 4; ++in)
        acc[im][in] = __builtin_amdgcn_wmma_f32_16x16x32_f16(
            false, a[im], false, b[in], (short)0, acc[im][in], false, false);

    if (it + 1 < K_ITERS) cvt_store(cur ^ 1);
  }

  // Epilogue: fp32 acc -> fp16 -> 16 MSB-first bit pulses as fp32.
  // C/D layout: VGPR g holds M = g + 8*(lane>>4), N = lane&15.
#pragma unroll
  for (int im = 0; im < 2; ++im) {
#pragma unroll
    for (int in = 0; in < 4; ++in) {
      const int n = bn + wn * 64 + in * 16 + lhalf;
#pragma unroll
      for (int g = 0; g < 8; ++g) {
        const int m = bm + wm * 32 + im * 16 + khi * 8 + g;
        union { _Float16 h; unsigned short u; } cv;
        cv.h = (_Float16)acc[im][in][g];
        const unsigned int u = cv.u;
        v4f* q = (v4f*)(out + ((size_t)m * N_DIM + n) * 16);
#pragma unroll
        for (int j = 0; j < 4; ++j) {
          v4f f;
          f[0] = (float)((u >> (15 - 4 * j)) & 1u);
          f[1] = (float)((u >> (14 - 4 * j)) & 1u);
          f[2] = (float)((u >> (13 - 4 * j)) & 1u);
          f[3] = (float)((u >> (12 - 4 * j)) & 1u);
          __builtin_nontemporal_store(f, q + j);   // NT: keep X/W resident in L2
        }
      }
    }
  }
}

extern "C" void kernel_launch(void* const* d_in, const int* in_sizes, int n_in,
                              void* d_out, int out_size, void* d_ws, size_t ws_size,
                              hipStream_t stream) {
  (void)n_in; (void)out_size; (void)d_ws; (void)ws_size;
  const float* X = (const float*)d_in[0];        // [M, K] fp32
  const float* W = (const float*)d_in[1];        // [N, K] fp32
  float* out = (float*)d_out;                    // [M, N, 16] fp32 pulses

  const int M = in_sizes[0] / K_DIM;             // 4096 in the reference
  dim3 grid(M / BM, N_DIM / BN);                 // (32, 16)
  spike_fp16_gemm_pulse<<<grid, dim3(256), 0, stream>>>(X, W, out);
}
